// QLinearPerGrpWithProfile_59425167507986
// MI455X (gfx1250) — matrix operand order repaired and checked by reading, not compiled, not run on hardware
//
#include <hip/hip_runtime.h>
#include <hip/hip_bf16.h>
#include <cstddef>

typedef __attribute__((ext_vector_type(16))) __bf16 v16bf;
typedef __attribute__((ext_vector_type(8)))  __bf16 v8bf;
typedef __attribute__((ext_vector_type(2)))  __bf16 v2bf;
typedef __attribute__((ext_vector_type(8)))  float  v8f;

#define IN_F    4096
#define OUT_F   11008
#define KSLICES 4
#define KS_LEN  (IN_F / KSLICES) /* 1024 */

// XOR-swizzled LDS offset: row-major [m][k] bf16, 1024 per row, with the
// 8-element K-block index xor'ed by (m&15) so that the 16 rows touched by one
// ds_load_b128 A-fragment read land on disjoint bank groups (no conflicts).
__device__ __forceinline__ int lds_off(int m, int k) {
  return m * KS_LEN + ((((k) >> 3) ^ (m & 15)) << 3) + (k & 7);
}

__global__ __launch_bounds__(256) void qlin4_init_bias(
    const float* __restrict__ bias, float* __restrict__ out) {
  int idx = blockIdx.x * 256 + threadIdx.x;
  if (idx < 32 * OUT_F) {
    out[idx] = (float)(__bf16)bias[idx % OUT_F];  // ref: bias -> bf16 -> f32
  }
}

__global__ __launch_bounds__(256) void qlin4_wmma(
    const float* __restrict__ x, const int* __restrict__ qw,
    const int* __restrict__ qzeros, const float* __restrict__ scales,
    float* __restrict__ out) {
  // 32 rows x 1024 K of x as bf16: exactly 64 KB of LDS.
  __shared__ __align__(16) __bf16 sx[32 * KS_LEN];

  const int tid    = threadIdx.x;
  const int kslice = blockIdx.y;
  const int kbase  = kslice * KS_LEN;

  // Stage the x K-slice to LDS (f32 -> bf16), 2 elements per thread per step:
  // coalesced float2 loads, packed b32 LDS stores.
  for (int i = tid * 2; i < 32 * KS_LEN; i += 512) {
    int m = i >> 10;
    int k = i & (KS_LEN - 1);
    float2 v = *(const float2*)&x[(size_t)m * IN_F + kbase + k];
    v2bf h;
    h[0] = (__bf16)v.x;
    h[1] = (__bf16)v.y;
    *(v2bf*)&sx[lds_off(m, k)] = h;
  }
  __syncthreads();

  const int wave  = tid >> 5;
  const int lane  = tid & 31;
  const int khalf = lane >> 4;   // B: K half (0..15 / 16..31); A: +16 K offset
  const int l15   = lane & 15;   // A: row m; B/out: column within tile
  const int n     = (blockIdx.x * 8 + wave) * 16 + l15;

  v8f acc0 = {};  // M rows 0..15
  v8f acc1 = {};  // M rows 16..31

  for (int g = 0; g < 8; ++g) {  // 128-wide quant groups inside this K slice
    const int   gi = kslice * 8 + g;
    const int   qz = qzeros[(size_t)gi * OUT_F + n];
    const float sc = (float)(__bf16)scales[(size_t)gi * OUT_F + n];
#pragma unroll
    for (int c = 0; c < 4; ++c) {  // four K=32 WMMA steps per group
      const int    k0 = g * 128 + c * 32;
      const size_t kq = (size_t)(kbase + k0 + khalf * 16);
      const int* qcol = qw + kq * OUT_F + n;

      // Prefetch next group's rows of this column (speculative, L2-bound).
      if (kbase + k0 + 159 < IN_F)
        __builtin_prefetch(qcol + (size_t)128 * OUT_F, 0, 0);

      // B fragment: lane holds 16 K-consecutive dequantized weights of col n.
      // (32x16 bf16 B: lanes 0-15 -> K 0..15, lanes 16-31 -> K 16..31.)
      v16bf b;
#pragma unroll
      for (int e = 0; e < 16; ++e) {
        int q = qcol[(size_t)e * OUT_F];            // global_load_b32, imm offs
        b[e] = (__bf16)((float)(q - qz) * sc);
      }

      // A fragments from LDS: 16x32 bf16 A, lane m = l15 (+16 for acc1),
      // elements 0..7 = K kb..kb+7, 8..15 = K kb+8..kb+15.
      const int kb = k0 + khalf * 16;
      v8bf a0lo = *(const v8bf*)&sx[lds_off(l15,      kb)];
      v8bf a0hi = *(const v8bf*)&sx[lds_off(l15,      kb + 8)];
      v8bf a1lo = *(const v8bf*)&sx[lds_off(l15 + 16, kb)];
      v8bf a1hi = *(const v8bf*)&sx[lds_off(l15 + 16, kb + 8)];
      v16bf a0, a1;
#pragma unroll
      for (int e = 0; e < 8; ++e) {
        a0[e] = a0lo[e]; a0[e + 8] = a0hi[e];
        a1[e] = a1lo[e]; a1[e + 8] = a1hi[e];
      }

      acc0 = __builtin_amdgcn_wmma_f32_16x16x32_bf16(
          false, a0, false, b, (short)0, acc0, false, false);
      acc1 = __builtin_amdgcn_wmma_f32_16x16x32_bf16(
          false, a1, false, b, (short)0, acc1, false, false);
    }
  }

  // Split-K reduction: accumulate partial tile into bias-initialized output.
  // C/D layout: lane -> N = l15, VGPR v -> M = v + khalf*8.
#pragma unroll
  for (int v = 0; v < 8; ++v) {
    const int m = v + khalf * 8;
    atomicAdd(&out[(size_t)m * OUT_F + n], acc0[v]);
    atomicAdd(&out[(size_t)(m + 16) * OUT_F + n], acc1[v]);
  }
}

extern "C" void kernel_launch(void* const* d_in, const int* in_sizes, int n_in,
                              void* d_out, int out_size, void* d_ws, size_t ws_size,
                              hipStream_t stream) {
  const float* x    = (const float*)d_in[0];
  const int*   qw   = (const int*)d_in[1];
  const int*   qz   = (const int*)d_in[2];
  const float* sc   = (const float*)d_in[3];
  const float* bias = (const float*)d_in[4];
  float*       out  = (float*)d_out;

  (void)d_ws; (void)ws_size; (void)in_sizes; (void)n_in; (void)out_size;

  qlin4_init_bias<<<(32 * OUT_F + 255) / 256, 256, 0, stream>>>(bias, out);

  dim3 grid(OUT_F / (16 * 8), KSLICES);  // 86 x 4
  qlin4_wmma<<<grid, 256, 0, stream>>>(x, qw, qz, sc, out);
}
